// myLoss_86277303042475
// MI455X (gfx1250) — compile-verified, matrix operand order checked by reading, not run on hardware
//
#include <hip/hip_runtime.h>
#include <hip/hip_bf16.h>
#include <math.h>

// ---- problem dims (match reference) ----
#define B_   2048
#define T_   100
#define TP_  112          // T padded to 7*16 (pad rows zeroed)
#define V_   20000
#define NVG  625          // V/32  (bf16 WMMA k-groups)
#define NVT  1250         // V/16
#define NBT  128          // B/16
#define NTT  7            // TP_/16
#define NCH  25           // V-contraction chunks (625 = 25*25) -> 22400 waves for occupancy
#define CSTEPS 25         // NVG/NCH
#define NWB  (NTT*NBT*NCH)  // 22400 waves for kB/kSH

typedef __attribute__((ext_vector_type(2)))  float  v2f;
typedef __attribute__((ext_vector_type(8)))  float  v8f;
typedef __attribute__((ext_vector_type(4)))  int    v4i;
typedef __attribute__((ext_vector_type(16))) __bf16 v16bf;

union BfPack { v4i i[2]; v16bf b; };

// ---------- preprocessing ----------

// K = exp(-lam*M), fp32 (kept for kA; K fits L2)
__global__ void kExpK(const float* __restrict__ M, const int* __restrict__ alpha,
                      float* __restrict__ K) {
  int i = blockIdx.x * blockDim.x + threadIdx.x;
  if (i >= T_ * V_) return;
  float lam = (float)(*alpha);
  K[i] = expf(-lam * M[i]);
}

// bf16 K and K*M in WMMA 16-bit A-matrix swizzled tiling:
// element index i = (vg*TP_ + t)*32 + e ; e = half*16 + j
// half0 covers v-offsets {0..7,16..23}, half1 covers {8..15,24..31}
__global__ void kSwz(const float* __restrict__ M, const int* __restrict__ alpha,
                     __hip_bfloat16* __restrict__ Kbf, __hip_bfloat16* __restrict__ KMbf) {
  int i = blockIdx.x * blockDim.x + threadIdx.x;
  if (i >= NVG * TP_ * 32) return;
  int e  = i & 31;
  int t  = (i >> 5) % TP_;
  int vg = (i >> 5) / TP_;
  float kv = 0.f, km = 0.f;
  if (t < T_) {
    int hl = e >> 4, j = e & 15;
    int voff = (hl == 0) ? (j < 8 ? j : j + 8) : (j < 8 ? j + 8 : j + 16);
    float lam = (float)(*alpha);
    float m = M[(size_t)t * V_ + vg * 32 + voff];
    kv = expf(-lam * m);
    km = kv * m;
  }
  Kbf[i]  = __float2bfloat16(kv);
  KMbf[i] = __float2bfloat16(km);
}

// bT[v][b] = bf16(doc_word[b][v]) via LDS-tiled transpose
__global__ void kTransB(const float* __restrict__ dw, __hip_bfloat16* __restrict__ bT) {
  __shared__ float tile[16][17];
  int vT = blockIdx.x, bTl = blockIdx.y;
  int tx = threadIdx.x & 15, ty = threadIdx.x >> 4;
  tile[ty][tx] = dw[(size_t)(bTl * 16 + ty) * V_ + vT * 16 + tx];
  __syncthreads();
  bT[(size_t)(vT * 16 + ty) * B_ + bTl * 16 + tx] = __float2bfloat16(tile[tx][ty]);
}

// aT[t][b] = doc_topic[b][t] (zero-padded rows t>=100)
__global__ void kAT(const float* __restrict__ dt, float* __restrict__ aT) {
  int i = blockIdx.x * blockDim.x + threadIdx.x;
  if (i >= TP_ * B_) return;
  int t = i / B_, b = i % B_;
  aT[i] = (t < T_) ? dt[(size_t)b * T_ + t] : 0.f;
}

__global__ void kUInit(float* __restrict__ u) {
  int i = blockIdx.x * blockDim.x + threadIdx.x;
  if (i >= TP_ * B_) return;
  u[i] = (i / B_ < T_) ? (1.0f / (float)T_) : 0.f;
}

// ---------- Sinkhorn iteration kernels ----------

// kA: v = b / (K^T u). One wave per 16x16 (V,B) tile; 25x v_wmma_f32_16x16x4_f32.
// Stores v in bf16, pre-swizzled into the WMMA 16-bit *B-matrix* register tiling
// (1 KB per 32x16 tile) so kB reads it with two coalesced b128 loads per lane.
__global__ void kA(const float* __restrict__ K, const float* __restrict__ u,
                   const __hip_bfloat16* __restrict__ bT, char* __restrict__ vswz) {
  int wid  = (blockIdx.x * blockDim.x + threadIdx.x) >> 5;
  int lane = threadIdx.x & 31;
  if (wid >= NVT * NBT) return;
  int vTile = wid % NVT, bTile = wid / NVT;
  int hl = lane >> 4, n = lane & 15;
  int v0 = vTile * 16, b0 = bTile * 16;

  const float* Kp = K + (size_t)(hl * 2) * V_ + v0 + n;  // A: m = lane&15 (v), k = hl*2+{0,1}
  const float* up = u + (size_t)(hl * 2) * B_ + b0 + n;  // B: n = lane&15 (b)
  v8f acc = {};
#pragma unroll
  for (int t0 = 0; t0 < T_; t0 += 4) {
    v2f a, b;
    a.x = Kp[(size_t)(t0 + 0) * V_];
    a.y = Kp[(size_t)(t0 + 1) * V_];
    b.x = up[(t0 + 0) * B_];
    b.y = up[(t0 + 1) * B_];
    acc = __builtin_amdgcn_wmma_f32_16x16x4_f32(false, a, false, b, (short)0, acc,
                                                false, false);
  }
#pragma unroll
  for (int r = 0; r < 8; ++r) {
    int vv = v0 + r + hl * 8;                      // C layout: row = r (+8 for hi half)
    float bb = __bfloat162float(bT[(size_t)vv * B_ + b0 + n]);
    float vval = bb / acc[r];
    int vg = vv >> 5, k = vv & 31;
    int seg = k >> 4;                 // 0: goes to consumer VGPR0-3, 1: VGPR4-7
    int half_t = (k >> 3) & 1;        // destination lane half
    int d  = (k & 7) >> 1;            // dword within 16B
    int lo = k & 1;                   // low/high bf16 in dword
    size_t off = ((size_t)(bTile * NVG + vg) << 10)
               + (size_t)(seg * 512 + (half_t * 16 + n) * 16 + d * 4 + lo * 2);
    *(__hip_bfloat16*)(vswz + off) = __float2bfloat16(vval);
  }
}

// kB: wpart[chunk] = (K @ v) partials. One wave per (tTile,bTile,chunk) 16x16 tile;
// 25x v_wmma_f32_16x16x32_bf16 with coalesced b128 loads of pre-swizzled K and v.
__global__ void kB(const char* __restrict__ Kbf, const char* __restrict__ vswz,
                   float* __restrict__ wpart) {
  int wid  = (blockIdx.x * blockDim.x + threadIdx.x) >> 5;
  int lane = threadIdx.x & 31;
  if (wid >= NWB) return;
  int chunk = wid / (NTT * NBT);
  int rem   = wid % (NTT * NBT);
  int tTile = rem % NTT, bTile = rem / NTT;
  int hl = lane >> 4, n = lane & 15;

  const char* ka0 = Kbf + (size_t)(tTile * 16 + n) * 64 + hl * 32;
  const char* vb0 = vswz + ((size_t)(bTile * NVG) << 10) + lane * 16;
  v8f acc = {};
#pragma unroll 5
  for (int s = 0; s < CSTEPS; ++s) {
    int vg = chunk * CSTEPS + s;
    const char* ka = ka0 + (size_t)vg * (TP_ * 64);
    const char* vb = vb0 + ((size_t)vg << 10);
    BfPack A, Bm;
    A.i[0]  = *(const v4i*)(ka);
    A.i[1]  = *(const v4i*)(ka + 16);
    Bm.i[0] = *(const v4i*)(vb);
    Bm.i[1] = *(const v4i*)(vb + 512);
    acc = __builtin_amdgcn_wmma_f32_16x16x32_bf16(false, A.b, false, Bm.b, (short)0,
                                                  acc, false, false);
  }
  float* wp = wpart + (size_t)chunk * (TP_ * B_) + (size_t)(tTile * 16) * B_ + bTile * 16 + n;
#pragma unroll
  for (int r = 0; r < 8; ++r) wp[(size_t)(r + hl * 8) * B_] = acc[r];
}

// kU: u = a / sum(wpart); padded rows forced to 0 (avoids 0/0 NaN)
__global__ void kU(const float* __restrict__ aT, const float* __restrict__ wpart,
                   float* __restrict__ u) {
  int i = blockIdx.x * blockDim.x + threadIdx.x;
  if (i >= TP_ * B_) return;
  float w = 0.f;
#pragma unroll
  for (int c = 0; c < NCH; ++c) w += wpart[(size_t)c * (TP_ * B_) + i];
  u[i] = (i / B_ < T_) ? aT[i] / w : 0.f;
}

// kSH: partials of sum_{t,b} u * ((K.*M) @ v); same tiling as kB + wave reduce.
__global__ void kSH(const char* __restrict__ KMbf, const char* __restrict__ vswz,
                    const float* __restrict__ u, float* __restrict__ shpart) {
  int wid  = (blockIdx.x * blockDim.x + threadIdx.x) >> 5;
  int lane = threadIdx.x & 31;
  if (wid >= NWB) return;
  int chunk = wid / (NTT * NBT);
  int rem   = wid % (NTT * NBT);
  int tTile = rem % NTT, bTile = rem / NTT;
  int hl = lane >> 4, n = lane & 15;

  const char* ka0 = KMbf + (size_t)(tTile * 16 + n) * 64 + hl * 32;
  const char* vb0 = vswz + ((size_t)(bTile * NVG) << 10) + lane * 16;
  v8f acc = {};
#pragma unroll 5
  for (int s = 0; s < CSTEPS; ++s) {
    int vg = chunk * CSTEPS + s;
    const char* ka = ka0 + (size_t)vg * (TP_ * 64);
    const char* vb = vb0 + ((size_t)vg << 10);
    BfPack A, Bm;
    A.i[0]  = *(const v4i*)(ka);
    A.i[1]  = *(const v4i*)(ka + 16);
    Bm.i[0] = *(const v4i*)(vb);
    Bm.i[1] = *(const v4i*)(vb + 512);
    acc = __builtin_amdgcn_wmma_f32_16x16x32_bf16(false, A.b, false, Bm.b, (short)0,
                                                  acc, false, false);
  }
  const float* up = u + (size_t)(tTile * 16) * B_ + bTile * 16 + n;
  float p = 0.f;
#pragma unroll
  for (int r = 0; r < 8; ++r) p += up[(size_t)(r + hl * 8) * B_] * acc[r];
#pragma unroll
  for (int o = 16; o > 0; o >>= 1) p += __shfl_down(p, o, 32);
  if (lane == 0) shpart[wid] = p;
}

// ---------- reconstruction loss ----------

// logits = doc_topic @ (1-M), fp32 WMMA 16x16x4, stored fp32 [B,V]
__global__ void kR1(const float* __restrict__ dt, const float* __restrict__ M,
                    float* __restrict__ logits) {
  int wid  = (blockIdx.x * blockDim.x + threadIdx.x) >> 5;
  int lane = threadIdx.x & 31;
  if (wid >= NVT * NBT) return;
  int vTile = wid % NVT, bTile = wid / NVT;
  int hl = lane >> 4, n = lane & 15;
  int v0 = vTile * 16, b0 = bTile * 16;

  const float* ap = dt + (size_t)(b0 + n) * T_ + hl * 2;   // A: m = lane&15 (b)
  const float* mp = M + (size_t)(hl * 2) * V_ + v0 + n;    // B: n = lane&15 (v)
  v8f acc = {};
#pragma unroll
  for (int t0 = 0; t0 < T_; t0 += 4) {
    v2f a, b;
    a.x = ap[t0];
    a.y = ap[t0 + 1];
    b.x = 1.0f - mp[(size_t)(t0 + 0) * V_];
    b.y = 1.0f - mp[(size_t)(t0 + 1) * V_];
    acc = __builtin_amdgcn_wmma_f32_16x16x4_f32(false, a, false, b, (short)0, acc,
                                                false, false);
  }
#pragma unroll
  for (int r = 0; r < 8; ++r)
    logits[(size_t)(b0 + r + hl * 8) * V_ + v0 + n] = acc[r];
}

// per-row logsumexp + weighted sums; one block per document
__global__ void kR2(const float* __restrict__ logits, const float* __restrict__ x,
                    float* __restrict__ recpart) {
  __shared__ float sm[256];
  int b = blockIdx.x, tid = threadIdx.x;
  const float* L = logits + (size_t)b * V_;
  const float* X = x + (size_t)b * V_;
  float m = -1e30f;
  for (int v = tid; v < V_; v += 256) m = fmaxf(m, L[v]);
  sm[tid] = m; __syncthreads();
  for (int o = 128; o > 0; o >>= 1) { if (tid < o) sm[tid] = fmaxf(sm[tid], sm[tid + o]); __syncthreads(); }
  m = sm[0]; __syncthreads();
  float se = 0.f, sx = 0.f, sxl = 0.f;
  for (int v = tid; v < V_; v += 256) {
    float l = L[v], xv = X[v];
    se += expf(l - m);
    sx += xv;
    sxl += xv * l;
  }
  sm[tid] = se; __syncthreads();
  for (int o = 128; o > 0; o >>= 1) { if (tid < o) sm[tid] += sm[tid + o]; __syncthreads(); }
  se = sm[0]; __syncthreads();
  sm[tid] = sx; __syncthreads();
  for (int o = 128; o > 0; o >>= 1) { if (tid < o) sm[tid] += sm[tid + o]; __syncthreads(); }
  sx = sm[0]; __syncthreads();
  sm[tid] = sxl; __syncthreads();
  for (int o = 128; o > 0; o >>= 1) { if (tid < o) sm[tid] += sm[tid + o]; __syncthreads(); }
  sxl = sm[0];
  if (tid == 0) recpart[b] = sxl - (m + logf(se)) * sx;
}

// final deterministic reduction -> (rec_loss, sh_loss, joint)
__global__ void kR3(const float* __restrict__ recpart, const float* __restrict__ shpart,
                    const int* __restrict__ rlw, float* __restrict__ out) {
  __shared__ float sm[256];
  int tid = threadIdx.x;
  float sr = 0.f;
  for (int i = tid; i < B_; i += 256) sr += recpart[i];
  sm[tid] = sr; __syncthreads();
  for (int o = 128; o > 0; o >>= 1) { if (tid < o) sm[tid] += sm[tid + o]; __syncthreads(); }
  sr = sm[0]; __syncthreads();
  float ss = 0.f;
  for (int i = tid; i < NWB; i += 256) ss += shpart[i];
  sm[tid] = ss; __syncthreads();
  for (int o = 128; o > 0; o >>= 1) { if (tid < o) sm[tid] += sm[tid + o]; __syncthreads(); }
  ss = sm[0];
  if (tid == 0) {
    float rec = -sr / (float)B_;
    float sh  = ss / (float)B_;
    out[0] = rec;
    out[1] = sh;
    out[2] = (float)(*rlw) * rec + sh;
  }
}

extern "C" void kernel_launch(void* const* d_in, const int* in_sizes, int n_in,
                              void* d_out, int out_size, void* d_ws, size_t ws_size,
                              hipStream_t stream) {
  const float* x   = (const float*)d_in[0];   // [B,V]
  const float* dt  = (const float*)d_in[1];   // doc_topic [B,T]
  const float* dw  = (const float*)d_in[2];   // doc_word  [B,V]
  const float* M   = (const float*)d_in[3];   // [T,V]
  const int* alpha = (const int*)d_in[5];     // sh_alpha
  const int* rlw   = (const int*)d_in[6];     // rec_loss_weight
  float* out = (float*)d_out;
  char* ws = (char*)d_ws;

  // workspace layout (bytes); logits aliases bT+vswz (used only after sinkhorn)
  const size_t oK   = 0;                                   //  8,000,000
  const size_t oKbf = oK   + (size_t)T_ * V_ * 4;          //  4,480,000
  const size_t oKM  = oKbf + (size_t)NVG * TP_ * 64;       //  4,480,000
  const size_t oAT  = oKM  + (size_t)NVG * TP_ * 64;       //    917,504
  const size_t oU   = oAT  + (size_t)TP_ * B_ * 4;         //    917,504
  const size_t oW   = oU   + (size_t)TP_ * B_ * 4;         // 22,937,600
  const size_t oSHP = oW   + (size_t)NCH * TP_ * B_ * 4;   //     89,600
  const size_t oRP  = oSHP + (size_t)NWB * 4;              //      8,192
  const size_t oBT  = oRP  + (size_t)B_ * 4;               // 81,920,000
  const size_t oVS  = oBT  + (size_t)V_ * B_ * 2;          // 81,920,000
  const size_t oLG  = oBT;                                 // 163,840,000 alias

  float* K  = (float*)(ws + oK);
  __hip_bfloat16* Kbf  = (__hip_bfloat16*)(ws + oKbf);
  __hip_bfloat16* KMbf = (__hip_bfloat16*)(ws + oKM);
  float* aT = (float*)(ws + oAT);
  float* u  = (float*)(ws + oU);
  float* wpart = (float*)(ws + oW);
  float* shpart = (float*)(ws + oSHP);
  float* recpart = (float*)(ws + oRP);
  __hip_bfloat16* bT = (__hip_bfloat16*)(ws + oBT);
  char* vswz = ws + oVS;
  float* logits = (float*)(ws + oLG);

  // --- preprocessing ---
  kExpK<<<(T_ * V_ + 255) / 256, 256, 0, stream>>>(M, alpha, K);
  kSwz<<<(NVG * TP_ * 32 + 255) / 256, 256, 0, stream>>>(M, alpha, Kbf, KMbf);
  kTransB<<<dim3(NVT, NBT), 256, 0, stream>>>(dw, bT);
  kAT<<<(TP_ * B_ + 255) / 256, 256, 0, stream>>>(dt, aT);
  kUInit<<<(TP_ * B_ + 255) / 256, 256, 0, stream>>>(u);

  // --- 50 Sinkhorn iterations ---
  for (int it = 0; it < 50; ++it) {
    kA<<<(NVT * NBT) / 8, 256, 0, stream>>>(K, u, bT, vswz);
    kB<<<NWB / 8, 256, 0, stream>>>((const char*)Kbf, vswz, wpart);
    kU<<<(TP_ * B_ + 255) / 256, 256, 0, stream>>>(aT, wpart, u);
  }
  // final v update + sinkhorn divergence partials
  kA<<<(NVT * NBT) / 8, 256, 0, stream>>>(K, u, bT, vswz);
  kSH<<<NWB / 8, 256, 0, stream>>>((const char*)KMbf, vswz, u, shpart);

  // --- reconstruction loss ---
  kR1<<<(NVT * NBT) / 8, 256, 0, stream>>>(dt, M, logits);
  kR2<<<B_, 256, 0, stream>>>(logits, x, recpart);
  kR3<<<1, 256, 0, stream>>>(recpart, shpart, rlw, out);
}
